// ClusteringLayer_63067299774668
// MI455X (gfx1250) — matrix operand order repaired and checked by reading, not compiled, not run on hardware
//
#include <hip/hip_runtime.h>
#include <hip/hip_bf16.h>

typedef __attribute__((ext_vector_type(16))) _Float16 v16h;
typedef __attribute__((ext_vector_type(8)))  _Float16 v8h;
typedef __attribute__((ext_vector_type(8)))  float    v8f;

#define K_CL 512
#define D_F  256
#define ROWS_PER_WG 64

__device__ __forceinline__ float dot4(float4 a) {
    return a.x * a.x + a.y * a.y + a.z * a.z + a.w * a.w;
}

__device__ __forceinline__ v8h cvt8(float4 a, float4 b) {
    v8h r;
    r[0] = (_Float16)a.x; r[1] = (_Float16)a.y; r[2] = (_Float16)a.z; r[3] = (_Float16)a.w;
    r[4] = (_Float16)b.x; r[5] = (_Float16)b.y; r[6] = (_Float16)b.z; r[7] = (_Float16)b.w;
    return r;
}

__device__ __forceinline__ v16h pack16(float4 a, float4 b, float4 c, float4 d) {
    v16h r;
    r[0]  = (_Float16)a.x; r[1]  = (_Float16)a.y; r[2]  = (_Float16)a.z; r[3]  = (_Float16)a.w;
    r[4]  = (_Float16)b.x; r[5]  = (_Float16)b.y; r[6]  = (_Float16)b.z; r[7]  = (_Float16)b.w;
    r[8]  = (_Float16)c.x; r[9]  = (_Float16)c.y; r[10] = (_Float16)c.z; r[11] = (_Float16)c.w;
    r[12] = (_Float16)d.x; r[13] = (_Float16)d.y; r[14] = (_Float16)d.z; r[15] = (_Float16)d.w;
    return r;
}

// LDS layout:
//   [0, 262144)            : packed f16 B tiles  [32 ct][8 ks][32 lane][16 halves]
//   [262144, +2048)        : c_sq[512]
//   [+2048, +256)          : rowsum[64]
#define SMEM_B_BYTES   (K_CL * D_F * 2)
#define SMEM_TOTAL     (SMEM_B_BYTES + K_CL * 4 + ROWS_PER_WG * 4)

__global__ __launch_bounds__(256) void cluster_q_kernel(
    const float* __restrict__ x, const float* __restrict__ clusters,
    float* __restrict__ out) {
    extern __shared__ char smem[];
    _Float16* smemB   = (_Float16*)smem;
    float*    csqL    = (float*)(smem + SMEM_B_BYTES);
    float*    rowsumL = csqL + K_CL;

    const int tid = threadIdx.x;
    if (tid < ROWS_PER_WG) rowsumL[tid] = 0.0f;

    // ---------------- Phase 1: pack clusters -> f16 WMMA-B layout in LDS ----
    // B k order used here (contiguous, same as A): lanes 0-15 hold K=0..15 (halves h=K),
    // lanes 16-31 hold K=16..31 (halves h=K-16); N = lane%16.
    #pragma unroll
    for (int rr = 0; rr < 2; ++rr) {
        const int n = tid * 2 + rr;            // cluster index 0..511
        const int Lmod = n & 15;
        const int ct = n >> 4;                 // column tile 0..31
        const float* crow = clusters + (size_t)n * D_F;
        float cs = 0.0f;
        #pragma unroll
        for (int ks = 0; ks < 8; ++ks) {
            const float4 f0 = *(const float4*)(crow + ks * 32 + 0);
            const float4 f1 = *(const float4*)(crow + ks * 32 + 4);
            const float4 f2 = *(const float4*)(crow + ks * 32 + 8);
            const float4 f3 = *(const float4*)(crow + ks * 32 + 12);
            const float4 g0 = *(const float4*)(crow + ks * 32 + 16);
            const float4 g1 = *(const float4*)(crow + ks * 32 + 20);
            const float4 g2 = *(const float4*)(crow + ks * 32 + 24);
            const float4 g3 = *(const float4*)(crow + ks * 32 + 28);
            cs += dot4(f0) + dot4(f1) + dot4(f2) + dot4(f3)
                + dot4(g0) + dot4(g1) + dot4(g2) + dot4(g3);
            _Float16* dst0 = smemB + (size_t)((ct * 8 + ks) * 32 + Lmod) * 16;
            *(v8h*)(dst0)     = cvt8(f0, f1);   // K = ks*32 + 0..7
            *(v8h*)(dst0 + 8) = cvt8(f2, f3);   // K = ks*32 + 8..15
            _Float16* dst1 = smemB + (size_t)((ct * 8 + ks) * 32 + Lmod + 16) * 16;
            *(v8h*)(dst1)     = cvt8(g0, g1);   // K = ks*32 + 16..23
            *(v8h*)(dst1 + 8) = cvt8(g2, g3);   // K = ks*32 + 24..31
        }
        csqL[n] = cs;
    }
    __syncthreads();

    // ---------------- Phase 2: WMMA GEMM over this block's 64 rows ----------
    // Waves 0-3: row-tiles {0,1}, column-tiles (w&3)*8 .. +7
    // Waves 4-7: row-tiles {2,3}, column-tiles (w&3)*8 .. +7
    const int w      = tid >> 5;
    const int w4     = w & 3;          // column-tile group: 8 tiles
    const int rowbase = (w >> 2) * 32; // 0 or 32 within the workgroup
    const int lane   = tid & 31;
    const int laneM  = lane & 15;
    const bool hi    = lane >= 16;
    const int sel    = hi ? 16 : 0;    // A layout: lane<16 K{0-15}, hi K{16-31}
    const size_t m0  = (size_t)blockIdx.x * ROWS_PER_WG;
    const float* xr0 = x + (m0 + (size_t)(rowbase + laneM)) * D_F;  // row tile A
    const float* xr1 = xr0 + 16 * D_F;                              // row tile B

    v8f acc[2][8];
    #pragma unroll
    for (int i = 0; i < 2; ++i)
        #pragma unroll
        for (int j = 0; j < 8; ++j) { v8f z = {}; acc[i][j] = z; }

    float xsq0 = 0.0f, xsq1 = 0.0f;
    #pragma unroll
    for (int ks = 0; ks < 8; ++ks) {
        const int dbase = ks * 32 + sel;
        const float4 p0 = *(const float4*)(xr0 + dbase);
        const float4 p1 = *(const float4*)(xr0 + dbase + 4);
        const float4 p2 = *(const float4*)(xr0 + dbase + 8);
        const float4 p3 = *(const float4*)(xr0 + dbase + 12);
        const float4 q0 = *(const float4*)(xr1 + dbase);
        const float4 q1 = *(const float4*)(xr1 + dbase + 4);
        const float4 q2 = *(const float4*)(xr1 + dbase + 8);
        const float4 q3 = *(const float4*)(xr1 + dbase + 12);
        xsq0 += dot4(p0) + dot4(p1) + dot4(p2) + dot4(p3);
        xsq1 += dot4(q0) + dot4(q1) + dot4(q2) + dot4(q3);
        const v16h a0 = pack16(p0, p1, p2, p3);
        const v16h a1 = pack16(q0, q1, q2, q3);
        #pragma unroll
        for (int j = 0; j < 8; ++j) {
            const v16h b = *(const v16h*)(smemB +
                (size_t)(((w4 * 8 + j) * 8 + ks) * 32 + lane) * 16);
            acc[0][j] = __builtin_amdgcn_wmma_f32_16x16x32_f16(
                false, a0, false, b, (short)0, acc[0][j], false, false);
            acc[1][j] = __builtin_amdgcn_wmma_f32_16x16x32_f16(
                false, a1, false, b, (short)0, acc[1][j], false, false);
        }
    }
    // lane L and L+16 hold complementary halves of row (L%16)'s |x|^2
    xsq0 += __shfl_xor(xsq0, 16, 32);
    xsq1 += __shfl_xor(xsq1, 16, 32);

    // ---------------- Epilogue: q = 1/(1+d^2), row sums ---------------------
    float rq0[8], rq1[8];
    #pragma unroll
    for (int r = 0; r < 8; ++r) { rq0[r] = 0.0f; rq1[r] = 0.0f; }

    #pragma unroll
    for (int j = 0; j < 8; ++j) {
        const float cn = csqL[(w4 * 8 + j) * 16 + laneM];
        #pragma unroll
        for (int r = 0; r < 8; ++r) {
            const int srcl = hi ? (r + 8) : r;    // C layout row for this lane/VGPR
            const float xsA = __shfl(xsq0, srcl, 32);
            float d2a = xsA + cn - 2.0f * acc[0][j][r];
            d2a = d2a > 0.0f ? d2a : 0.0f;
            const float qa = 1.0f / (1.0f + d2a);
            acc[0][j][r] = qa; rq0[r] += qa;

            const float xsB = __shfl(xsq1, srcl, 32);
            float d2b = xsB + cn - 2.0f * acc[1][j][r];
            d2b = d2b > 0.0f ? d2b : 0.0f;
            const float qb = 1.0f / (1.0f + d2b);
            acc[1][j][r] = qb; rq1[r] += qb;
        }
    }
    #pragma unroll
    for (int r = 0; r < 8; ++r) {
        float s0 = rq0[r], s1 = rq1[r];
        s0 += __shfl_xor(s0, 1, 32); s0 += __shfl_xor(s0, 2, 32);
        s0 += __shfl_xor(s0, 4, 32); s0 += __shfl_xor(s0, 8, 32);
        s1 += __shfl_xor(s1, 1, 32); s1 += __shfl_xor(s1, 2, 32);
        s1 += __shfl_xor(s1, 4, 32); s1 += __shfl_xor(s1, 8, 32);
        if (laneM == 0) {
            const int row = hi ? (r + 8) : r;
            atomicAdd(&rowsumL[rowbase + row], s0);
            atomicAdd(&rowsumL[rowbase + 16 + row], s1);
        }
    }
    __syncthreads();

    // ---------------- Normalize + store -------------------------------------
    float* outb = out + m0 * K_CL;
    #pragma unroll
    for (int r = 0; r < 8; ++r) {
        const int row0 = rowbase + (hi ? (r + 8) : r);
        const float inv0 = 1.0f / rowsumL[row0];
        const float inv1 = 1.0f / rowsumL[row0 + 16];
        #pragma unroll
        for (int j = 0; j < 8; ++j) {
            const int col = (w4 * 8 + j) * 16 + laneM;
            outb[(size_t)row0 * K_CL + col]        = acc[0][j][r] * inv0;
            outb[(size_t)(row0 + 16) * K_CL + col] = acc[1][j][r] * inv1;
        }
    }
}

extern "C" void kernel_launch(void* const* d_in, const int* in_sizes, int n_in,
                              void* d_out, int out_size, void* d_ws, size_t ws_size,
                              hipStream_t stream) {
    (void)n_in; (void)out_size; (void)d_ws; (void)ws_size;
    const float* x        = (const float*)d_in[0];
    const float* clusters = (const float*)d_in[1];
    float* out = (float*)d_out;
    const int N = in_sizes[0] / D_F;          // 262144
    const int grid = N / ROWS_PER_WG;         // 64 rows per workgroup
    cluster_q_kernel<<<grid, 256, SMEM_TOTAL, stream>>>(x, clusters, out);
}
